// GNN_47339129536979
// MI455X (gfx1250) — matrix operand order repaired and checked by reading, not compiled, not run on hardware
//
#include <hip/hip_runtime.h>

// ---------------------------------------------------------------------------
// GNN forward for MI455X (gfx1250): bf16 WMMA (16x16x32) with f32 accum.
// Shapes: N=10000 nodes, E=320000 edges, HID=128, NEF=142 (padded 160), 4 layers.
// ---------------------------------------------------------------------------

#define NNODE 10000
#define NEDGE 320000
#define HID   128
#define NEF_  142
#define NLAY  4

typedef __attribute__((ext_vector_type(16))) __bf16 bf16x16;
typedef __attribute__((ext_vector_type(8)))  __bf16 bf16x8;
typedef __attribute__((ext_vector_type(8)))  float  f32x8;
typedef int v4i_ __attribute__((vector_size(16)));

__device__ __forceinline__ f32x8 wmma_bf16(bf16x16 a, bf16x16 b, f32x8 c) {
  // D = A(16x32) * B(32x16) + C, f32 accumulate
  return __builtin_amdgcn_wmma_f32_16x16x32_bf16(false, a, false, b, (short)0, c,
                                                 false, false);
}

__device__ __forceinline__ float silu_f(float v) {
  return v * (1.0f / (1.0f + __expf(-v)));
}

// Wave-internal LDS RAW fence (store by lane i, load by lane j).
__device__ __forceinline__ void wave_lds_fence() {
#if __has_builtin(__builtin_amdgcn_s_wait_dscnt)
  __builtin_amdgcn_s_wait_dscnt(0);
#else
  asm volatile("s_wait_dscnt 0" ::: "memory");
#endif
}

// 16B global -> LDS copy, async (gfx1250 builtin takes int4 pointers:
// (v4i AS1* gsrc, v4i AS3* lds, imm offset, imm cpol)).
__device__ __forceinline__ void g2s16(const void* g, void* s) {
#if __has_builtin(__builtin_amdgcn_global_load_async_to_lds_b128)
  __builtin_amdgcn_global_load_async_to_lds_b128(
      (__attribute__((address_space(1))) v4i_*)g,
      (__attribute__((address_space(3))) v4i_*)s, 0, 0);
#else
  *(uint4*)s = *(const uint4*)g;
#endif
}
__device__ __forceinline__ void g2s_fence() {
#if __has_builtin(__builtin_amdgcn_s_wait_asynccnt)
  __builtin_amdgcn_s_wait_asynccnt(0);
#elif __has_builtin(__builtin_amdgcn_global_load_async_to_lds_b128)
  asm volatile("s_wait_asynccnt 0" ::: "memory");
#endif
}

// A fragment (16x32 bf16, MxK) from LDS row-major rows (stride in elems).
// lane<16: row=lane, K = kb+0..7 (elems 0..7), kb+16..23 (elems 8..15)
// lane>=16: row=lane-16, K = kb+8..15, kb+24..31
__device__ __forceinline__ bf16x16 load_afrag(const __bf16* base, int stride,
                                              int kb, int lane) {
  const __bf16* p = base + (lane & 15) * stride + kb + ((lane >> 4) << 3);
  bf16x8 lo = *(const bf16x8*)p;
  bf16x8 hi = *(const bf16x8*)(p + 16);
  return __builtin_shufflevector(lo, hi, 0, 1, 2, 3, 4, 5, 6, 7,
                                 8, 9, 10, 11, 12, 13, 14, 15);
}

// B fragment: prepacked contiguous 512-elem tile, lane takes 16 consecutive.
__device__ __forceinline__ bf16x16 load_bfrag(const __bf16* tile, int lane) {
  return *(const bf16x16*)(tile + lane * 16);
}

// ---------------------------------------------------------------------------
// Weight prepack: f32 row-major W[K x Nreal] -> bf16 B-fragment layout
// dst[((ks*ntiles)+nt)*512 + lane*16 + j] with
//   n = nt*16 + (lane&15),  k = ks*32 + ((lane>=16)?8:0) + (j&7) + ((j>=8)?16:0)
// Zero-pads K -> Kp and Nreal -> Nout.
// ---------------------------------------------------------------------------
__global__ void k_prepack(const float* __restrict__ W, __bf16* __restrict__ dst,
                          int K, int Kp, int Nout, int Nreal) {
  int ntiles = Nout >> 4;
  int total  = (Kp >> 5) * ntiles * 512;
  for (int i = blockIdx.x * blockDim.x + threadIdx.x; i < total;
       i += gridDim.x * blockDim.x) {
    int j    = i & 15;
    int lane = (i >> 4) & 31;
    int t    = i >> 9;
    int nt   = t % ntiles;
    int ks   = t / ntiles;
    int n = nt * 16 + (lane & 15);
    int k = (ks << 5) + ((lane >> 4) << 3) + (j & 7) + ((j >> 3) << 4);
    float v = 0.0f;
    if (k < K && n < Nreal) v = W[(long)k * Nreal + n];
    dst[i] = (__bf16)v;
  }
}

__global__ void k_copy(float* __restrict__ d, const float* __restrict__ s, int n) {
  int i = blockIdx.x * blockDim.x + threadIdx.x;
  if (i < n) d[i] = s[i];
}
__global__ void k_zero(float* __restrict__ d, int n) {
  int i = blockIdx.x * blockDim.x + threadIdx.x;
  if (i < n) d[i] = 0.0f;
}
__global__ void k_count(const int* __restrict__ recv, float* __restrict__ c, int n) {
  int i = blockIdx.x * blockDim.x + threadIdx.x;
  if (i < n) atomicAdd(&c[recv[i]], 1.0f);
}
__global__ void k_clamp(float* __restrict__ c, int n) {
  int i = blockIdx.x * blockDim.x + threadIdx.x;
  if (i < n) c[i] = fmaxf(c[i], 1.0f);
}

// ---------------------------------------------------------------------------
// Edge MLP: m = SiLU(SiLU(ea @ W1 + b1) @ W2 + b2); scatter-add into agg[recv].
// Block = 256 threads (8 waves), 16 edges per wave, 128 edges per block.
// layer0: ea = edge_attr (K=142 padded to Kp=160)
// else  : ea = concat(x[send], x[recv], m_prev)  (Kp=384)
// ---------------------------------------------------------------------------
__global__ __launch_bounds__(256) void k_edge(
    const float* __restrict__ xbuf, const float* __restrict__ edge_attr,
    float* __restrict__ mbuf, const int* __restrict__ send,
    const int* __restrict__ recv, const __bf16* __restrict__ pW1,
    const float* __restrict__ b1, const __bf16* __restrict__ pW2,
    const float* __restrict__ b2, float* __restrict__ agg, int Kp, int layer0) {
  extern __shared__ __attribute__((aligned(16))) char smem[];
  const int nks1 = Kp >> 5;
  __bf16* sW1 = (__bf16*)smem;           // nks1*8*512 bf16
  __bf16* sW2 = sW1 + nks1 * 4096;       // 4*8*512 = 16384 bf16
  __bf16* sA  = sW2 + 16384;             // 8 waves * 16 rows * Kp bf16

  const int tid   = threadIdx.x;
  const int wave  = tid >> 5;
  const int lane  = tid & 31;
  const int col0  = lane & 15;
  const int hi    = lane >> 4;
  const int eBase = blockIdx.x * 128;

  // Stage packed weights (contiguous, 16B per lane, async to LDS).
  for (int i = tid; i < nks1 * 512; i += 256) g2s16(pW1 + i * 8, sW1 + i * 8);
  for (int i = tid; i < 2048; i += 256)       g2s16(pW2 + i * 8, sW2 + i * 8);

  // Stage A rows (bf16) for this wave's 16 edges, coalesced per row.
  {
    __bf16* dstW = sA + wave * 16 * Kp;
    for (int r = 0; r < 16; ++r) {
      const int e = eBase + wave * 16 + r;
      __bf16* dst = dstW + r * Kp;
      if (layer0) {
        for (int k = lane; k < Kp; k += 32)
          dst[k] = (__bf16)((k < NEF_) ? edge_attr[(long)e * NEF_ + k] : 0.0f);
      } else {
        const float* ps = xbuf + (long)send[e] * HID;
        const float* pr = xbuf + (long)recv[e] * HID;
        const float* pm = mbuf + (long)e * HID;
        for (int k = lane; k < HID; k += 32) {
          dst[k]           = (__bf16)ps[k];
          dst[HID + k]     = (__bf16)pr[k];
          dst[2 * HID + k] = (__bf16)pm[k];
        }
      }
    }
  }
  g2s_fence();
  __syncthreads();

  const f32x8 zero8 = {0, 0, 0, 0, 0, 0, 0, 0};

  // GEMM1: [16 x Kp] @ [Kp x 128]
  const __bf16* aw = sA + wave * 16 * Kp;
  f32x8 acc[8];
#pragma unroll
  for (int nt = 0; nt < 8; ++nt) acc[nt] = zero8;
  for (int ks = 0; ks < nks1; ++ks) {
    bf16x16 a = load_afrag(aw, Kp, ks << 5, lane);
    const __bf16* bb = sW1 + ks * 4096;
#pragma unroll
    for (int nt = 0; nt < 8; ++nt)
      acc[nt] = wmma_bf16(a, load_bfrag(bb + nt * 512, lane), acc[nt]);
  }

  // hidden = SiLU(acc + b1) -> bf16 rows (stride 128), reuse wave's A region.
  __bf16* hw = sA + wave * 16 * Kp;
#pragma unroll
  for (int nt = 0; nt < 8; ++nt) {
    const float b = b1[nt * 16 + col0];
#pragma unroll
    for (int r = 0; r < 8; ++r) {
      float v = silu_f(acc[nt][r] + b);  // C layout: row M = r + hi*8, col = nt*16+col0
      hw[(r + hi * 8) * HID + nt * 16 + col0] = (__bf16)v;
    }
  }
  wave_lds_fence();

  // GEMM2: [16 x 128] @ [128 x 128]
  f32x8 acc2[8];
#pragma unroll
  for (int nt = 0; nt < 8; ++nt) acc2[nt] = zero8;
#pragma unroll
  for (int ks = 0; ks < 4; ++ks) {
    bf16x16 a = load_afrag(hw, HID, ks << 5, lane);
    const __bf16* bb = sW2 + ks * 4096;
#pragma unroll
    for (int nt = 0; nt < 8; ++nt)
      acc2[nt] = wmma_bf16(a, load_bfrag(bb + nt * 512, lane), acc2[nt]);
  }

  // m = SiLU(acc2 + b2): store + scatter-mean numerator via f32 atomics.
#pragma unroll
  for (int r = 0; r < 8; ++r) {
    const int e  = eBase + wave * 16 + r + hi * 8;
    const int rv = recv[e];
#pragma unroll
    for (int nt = 0; nt < 8; ++nt) {
      float v = silu_f(acc2[nt][r] + b2[nt * 16 + col0]);
      mbuf[(long)e * HID + nt * 16 + col0] = v;
      atomicAdd(&agg[(long)rv * HID + nt * 16 + col0], v);
    }
  }
}

// ---------------------------------------------------------------------------
// Node update: xn = x + agg/count;  x' = xn + SiLU(xn@U1+ub1)@U2 + ub2
// ---------------------------------------------------------------------------
__global__ __launch_bounds__(256) void k_node(
    float* __restrict__ xbuf, const float* __restrict__ agg,
    const float* __restrict__ counts, const __bf16* __restrict__ pU1,
    const float* __restrict__ ub1, const __bf16* __restrict__ pU2,
    const float* __restrict__ ub2) {
  extern __shared__ __attribute__((aligned(16))) char smem[];
  __bf16* sU1 = (__bf16*)smem;          // 4*16*512 = 32768 bf16
  __bf16* sU2 = sU1 + 32768;            // 8*8*512  = 32768 bf16
  float*  sX  = (float*)(sU2 + 32768);  // 128*128 f32
  __bf16* sA  = (__bf16*)(sX + 16384);  // 128*128 bf16
  __bf16* sH  = sA + 16384;             // 128*256 bf16

  const int tid = threadIdx.x, wave = tid >> 5, lane = tid & 31;
  const int col0 = lane & 15, hi = lane >> 4;
  const int base = blockIdx.x * 128;

  for (int i = tid; i < 4096; i += 256) g2s16(pU1 + i * 8, sU1 + i * 8);
  for (int i = tid; i < 4096; i += 256) g2s16(pU2 + i * 8, sU2 + i * 8);

  for (int i = tid; i < 128 * HID; i += 256) {
    int nl = i >> 7, k = i & 127;
    int node = base + nl;
    float v = 0.0f;
    if (node < NNODE)
      v = xbuf[(long)node * HID + k] + agg[(long)node * HID + k] / counts[node];
    sX[i] = v;
    sA[i] = (__bf16)v;  // wave w owns rows [w*16, w*16+16): contiguous region
  }
  g2s_fence();
  __syncthreads();

  const f32x8 zero8 = {0, 0, 0, 0, 0, 0, 0, 0};

  // GEMM1: [16 x 128] @ [128 x 256]
  const __bf16* aw = sA + wave * 16 * HID;
  f32x8 acc[16];
#pragma unroll
  for (int nt = 0; nt < 16; ++nt) acc[nt] = zero8;
#pragma unroll
  for (int ks = 0; ks < 4; ++ks) {
    bf16x16 a = load_afrag(aw, HID, ks << 5, lane);
#pragma unroll
    for (int nt = 0; nt < 16; ++nt)
      acc[nt] = wmma_bf16(a, load_bfrag(sU1 + (ks * 16 + nt) * 512, lane), acc[nt]);
  }
  __bf16* hw = sH + wave * 16 * 256;
#pragma unroll
  for (int nt = 0; nt < 16; ++nt) {
    const float b = ub1[nt * 16 + col0];
#pragma unroll
    for (int r = 0; r < 8; ++r)
      hw[(r + hi * 8) * 256 + nt * 16 + col0] = (__bf16)silu_f(acc[nt][r] + b);
  }
  wave_lds_fence();

  // GEMM2: [16 x 256] @ [256 x 128]
  f32x8 acc2[8];
#pragma unroll
  for (int nt = 0; nt < 8; ++nt) acc2[nt] = zero8;
#pragma unroll
  for (int ks = 0; ks < 8; ++ks) {
    bf16x16 a = load_afrag(hw, 256, ks << 5, lane);
#pragma unroll
    for (int nt = 0; nt < 8; ++nt)
      acc2[nt] = wmma_bf16(a, load_bfrag(sU2 + (ks * 8 + nt) * 512, lane), acc2[nt]);
  }
#pragma unroll
  for (int r = 0; r < 8; ++r) {
    const int nl = wave * 16 + r + hi * 8;
    const int node = base + nl;
    if (node < NNODE) {
#pragma unroll
      for (int nt = 0; nt < 8; ++nt) {
        float v = sX[nl * HID + nt * 16 + col0] + acc2[nt][r] + ub2[nt * 16 + col0];
        xbuf[(long)node * HID + nt * 16 + col0] = v;
      }
    }
  }
}

// ---------------------------------------------------------------------------
// Output head: out = SiLU(SiLU(x@O1+ob1)@O2+ob2)@O3 + ob3   [N,3] (N padded 16)
// ---------------------------------------------------------------------------
__global__ __launch_bounds__(256) void k_out(
    const float* __restrict__ xbuf, const __bf16* __restrict__ pO1,
    const float* __restrict__ ob1, const __bf16* __restrict__ pO2,
    const float* __restrict__ ob2, const __bf16* __restrict__ pO3,
    const float* __restrict__ ob3, float* __restrict__ out) {
  extern __shared__ __attribute__((aligned(16))) char smem[];
  __bf16* sO1 = (__bf16*)smem;   // 16384
  __bf16* sO2 = sO1 + 16384;     // 16384
  __bf16* sO3 = sO2 + 16384;     // 2048
  __bf16* sA  = sO3 + 2048;      // 16384
  __bf16* sH  = sA + 16384;      // 16384

  const int tid = threadIdx.x, wave = tid >> 5, lane = tid & 31;
  const int col0 = lane & 15, hi = lane >> 4;
  const int base = blockIdx.x * 128;

  for (int i = tid; i < 2048; i += 256) g2s16(pO1 + i * 8, sO1 + i * 8);
  for (int i = tid; i < 2048; i += 256) g2s16(pO2 + i * 8, sO2 + i * 8);
  for (int i = tid; i < 256; i += 256)  g2s16(pO3 + i * 8, sO3 + i * 8);

  for (int i = tid; i < 128 * HID; i += 256) {
    int nl = i >> 7, k = i & 127;
    int node = base + nl;
    sA[i] = (__bf16)((node < NNODE) ? xbuf[(long)node * HID + k] : 0.0f);
  }
  g2s_fence();
  __syncthreads();

  const f32x8 zero8 = {0, 0, 0, 0, 0, 0, 0, 0};
  __bf16* aw = sA + wave * 16 * HID;
  __bf16* hw = sH + wave * 16 * HID;

  // h1 = SiLU(x@O1 + ob1)
  f32x8 acc[8];
#pragma unroll
  for (int nt = 0; nt < 8; ++nt) acc[nt] = zero8;
#pragma unroll
  for (int ks = 0; ks < 4; ++ks) {
    bf16x16 a = load_afrag(aw, HID, ks << 5, lane);
#pragma unroll
    for (int nt = 0; nt < 8; ++nt)
      acc[nt] = wmma_bf16(a, load_bfrag(sO1 + (ks * 8 + nt) * 512, lane), acc[nt]);
  }
#pragma unroll
  for (int nt = 0; nt < 8; ++nt) {
    const float b = ob1[nt * 16 + col0];
#pragma unroll
    for (int r = 0; r < 8; ++r)
      hw[(r + hi * 8) * HID + nt * 16 + col0] = (__bf16)silu_f(acc[nt][r] + b);
  }
  wave_lds_fence();

  // h2 = SiLU(h1@O2 + ob2)  (write back into wave's sA region)
#pragma unroll
  for (int nt = 0; nt < 8; ++nt) acc[nt] = zero8;
#pragma unroll
  for (int ks = 0; ks < 4; ++ks) {
    bf16x16 a = load_afrag(hw, HID, ks << 5, lane);
#pragma unroll
    for (int nt = 0; nt < 8; ++nt)
      acc[nt] = wmma_bf16(a, load_bfrag(sO2 + (ks * 8 + nt) * 512, lane), acc[nt]);
  }
#pragma unroll
  for (int nt = 0; nt < 8; ++nt) {
    const float b = ob2[nt * 16 + col0];
#pragma unroll
    for (int r = 0; r < 8; ++r)
      aw[(r + hi * 8) * HID + nt * 16 + col0] = (__bf16)silu_f(acc[nt][r] + b);
  }
  wave_lds_fence();

  // out = h2@O3 + ob3 (N padded to 16, only cols 0..2 stored)
  f32x8 acc3 = zero8;
#pragma unroll
  for (int ks = 0; ks < 4; ++ks) {
    bf16x16 a = load_afrag(aw, HID, ks << 5, lane);
    acc3 = wmma_bf16(a, load_bfrag(sO3 + ks * 512, lane), acc3);
  }
  if (col0 < 3) {
    const float b = ob3[col0];
#pragma unroll
    for (int r = 0; r < 8; ++r) {
      const int node = base + wave * 16 + r + hi * 8;
      if (node < NNODE) out[(long)node * 3 + col0] = acc3[r] + b;
    }
  }
}

// ---------------------------------------------------------------------------
// Host launcher
// ---------------------------------------------------------------------------
extern "C" void kernel_launch(void* const* d_in, const int* in_sizes, int n_in,
                              void* d_out, int out_size, void* d_ws,
                              size_t ws_size, hipStream_t stream) {
  (void)in_sizes; (void)n_in; (void)out_size; (void)ws_size;

  // Inputs in setup_inputs() dict insertion order.
  const float* x    = (const float*)d_in[0];
  const float* ea   = (const float*)d_in[1];
  const int*   send = (const int*)d_in[2];
  const int*   recv = (const int*)d_in[3];
  const float *mw1[NLAY], *mb1[NLAY], *mw2[NLAY], *mb2[NLAY];
  const float *uw1[NLAY], *ub1[NLAY], *uw2[NLAY], *ub2[NLAY];
  int idx = 4;
  for (int i = 0; i < NLAY; ++i) {
    mw1[i] = (const float*)d_in[idx++]; mb1[i] = (const float*)d_in[idx++];
    mw2[i] = (const float*)d_in[idx++]; mb2[i] = (const float*)d_in[idx++];
    uw1[i] = (const float*)d_in[idx++]; ub1[i] = (const float*)d_in[idx++];
    uw2[i] = (const float*)d_in[idx++]; ub2[i] = (const float*)d_in[idx++];
  }
  const float* ow1 = (const float*)d_in[idx++];
  const float* ob1 = (const float*)d_in[idx++];
  const float* ow2 = (const float*)d_in[idx++];
  const float* ob2 = (const float*)d_in[idx++];
  const float* ow3 = (const float*)d_in[idx++];
  const float* ob3 = (const float*)d_in[idx++];

  // Workspace carve-up (256B aligned).
  char* cur = (char*)d_ws;
  auto take = [&](size_t bytes) -> char* {
    char* p = cur;
    cur += (bytes + 255) & ~(size_t)255;
    return p;
  };
  float* xbuf = (float*)take((size_t)NNODE * HID * 4);
  float* mbuf = (float*)take((size_t)NEDGE * HID * 4);
  float* agg  = (float*)take((size_t)NNODE * HID * 4);
  float* cnts = (float*)take((size_t)NNODE * 4);
  __bf16 *pmw1[NLAY], *pmw2[NLAY], *puw1[NLAY], *puw2[NLAY];
  for (int i = 0; i < NLAY; ++i) {
    int Kp = (i == 0) ? 160 : 384;
    pmw1[i] = (__bf16*)take((size_t)(Kp / 32) * 8 * 512 * 2);
    pmw2[i] = (__bf16*)take((size_t)4 * 8 * 512 * 2);
    puw1[i] = (__bf16*)take((size_t)4 * 16 * 512 * 2);
    puw2[i] = (__bf16*)take((size_t)8 * 8 * 512 * 2);
  }
  __bf16* pow1 = (__bf16*)take((size_t)4 * 8 * 512 * 2);
  __bf16* pow2 = (__bf16*)take((size_t)4 * 8 * 512 * 2);
  __bf16* pow3 = (__bf16*)take((size_t)4 * 1 * 512 * 2);

  // Init: x copy, in-degree counts (clamped to >= 1).
  k_copy<<<(NNODE * HID + 255) / 256, 256, 0, stream>>>(xbuf, x, NNODE * HID);
  k_zero<<<(NNODE + 255) / 256, 256, 0, stream>>>(cnts, NNODE);
  k_count<<<(NEDGE + 255) / 256, 256, 0, stream>>>(recv, cnts, NEDGE);
  k_clamp<<<(NNODE + 255) / 256, 256, 0, stream>>>(cnts, NNODE);

  // Weight prepack (f32 -> bf16 B-fragment layout).
  auto pp = [&](const float* w, __bf16* dst, int K, int Kp, int Nout, int Nreal) {
    int total = (Kp / 32) * (Nout / 16) * 512;
    k_prepack<<<(total + 255) / 256, 256, 0, stream>>>(w, dst, K, Kp, Nout, Nreal);
  };
  for (int i = 0; i < NLAY; ++i) {
    pp(mw1[i], pmw1[i], (i == 0) ? NEF_ : 384, (i == 0) ? 160 : 384, 128, 128);
    pp(mw2[i], pmw2[i], 128, 128, 128, 128);
    pp(uw1[i], puw1[i], 128, 128, 256, 256);
    pp(uw2[i], puw2[i], 256, 256, 128, 128);
  }
  pp(ow1, pow1, 128, 128, 128, 128);
  pp(ow2, pow2, 128, 128, 128, 128);
  pp(ow3, pow3, 128, 128, 16, 3);

  // Layers.
  for (int l = 0; l < NLAY; ++l) {
    k_zero<<<(NNODE * HID + 255) / 256, 256, 0, stream>>>(agg, NNODE * HID);
    int Kp = (l == 0) ? 160 : 384;
    size_t smem_edge =
        (size_t)(Kp / 32) * 8192 + 32768 + (size_t)128 * Kp * 2;  // W1 + W2 + A
    k_edge<<<NEDGE / 128, 256, smem_edge, stream>>>(
        xbuf, ea, mbuf, send, recv, pmw1[l], mb1[l], pmw2[l], mb2[l], agg, Kp,
        (l == 0) ? 1 : 0);
    k_node<<<(NNODE + 127) / 128, 256, 294912, stream>>>(
        xbuf, agg, cnts, puw1[l], ub1[l], puw2[l], ub2[l]);
  }

  // Output head.
  k_out<<<(NNODE + 127) / 128, 256, 135168, stream>>>(
      xbuf, pow1, ob1, pow2, ob2, pow3, ob3, (float*)d_out);
}
